// CRFVGG_prune_29935922053302
// MI455X (gfx1250) — matrix operand order repaired and checked by reading, compiled, not run on hardware
//
#include <hip/hip_runtime.h>
#include <stdint.h>

typedef __attribute__((ext_vector_type(16))) __bf16 v16bf;
typedef __attribute__((ext_vector_type(8)))  __bf16 v8bf;
typedef __attribute__((ext_vector_type(8)))  float  v8f;

union FragBF {
    v16bf v;
    v8bf  h[2];
};

#define HW 65536      // 256*256 pixels per (batch, channel) plane
#define NCH_TOT 390   // 228 + 111 + 51

// ---------------------------------------------------------------------------
// Prologue 1: pre-swizzle weights into WMMA A-fragment layout (bf16 hi + lo).
// Record = (mtile16, kstep): 32 lanes x (16 hi + 16 lo bf16) = 2048 bytes.
// A-fragment element mapping for V_WMMA_F32_16X16X32_BF16 (16-bit A 16x32):
//   lane L: M = mtile*16 + (L%16), half = L/16
//   element e (0..15): v = e/2, K = kstep*32 + (v<4?0:16) + half*8 + (v&3)*2 + (e&1)
// ---------------------------------------------------------------------------
__global__ void prep_weights(const float* __restrict__ wA,  // [C][C1]
                             const float* __restrict__ wB,  // [C][C2]
                             int C, int C1, int C2, int ksteps,
                             __bf16* __restrict__ out)      // branch record base
{
    int rec   = blockIdx.x;           // mtile*ksteps + kstep
    int mtile = rec / ksteps;
    int kstep = rec % ksteps;
    int lane  = threadIdx.x;          // 0..31
    int m     = mtile * 16 + (lane & 15);
    int half  = lane >> 4;

    __bf16* dst = out + (size_t)rec * 1024 + (size_t)lane * 32;

#pragma unroll
    for (int e = 0; e < 16; ++e) {
        int v = e >> 1;
        int k = kstep * 32 + ((v < 4) ? 0 : 16) + half * 8 + (v & 3) * 2 + (e & 1);
        float val = 0.0f;
        if (m < C) {
            if (k < C1)            val = wA[(size_t)m * C1 + k];
            else if (k < C1 + C2)  val = wB[(size_t)m * C2 + (k - C1)];
        }
        __bf16 hi = (__bf16)val;
        __bf16 lo = (__bf16)(val - (float)hi);
        dst[e]      = hi;
        dst[16 + e] = lo;
    }
}

// Prologue 2: per-destination bias = sum of the two incoming edge biases.
__global__ void prep_bias(const float* __restrict__ bA,
                          const float* __restrict__ bB,
                          int C, float* __restrict__ out)
{
    int m = blockIdx.x * blockDim.x + threadIdx.x;
    if (m < C) out[m] = bA[m] + bB[m];
}

// ---------------------------------------------------------------------------
// Main fused kernel: one message-passing iteration for all 3 branches.
// grid = (7 m-blocks across branches, 512 n-tiles of 128 px, 2 batches)
//   -> x varies fastest so all 7 m-blocks of one pixel window run together
//      and share the activation stripe through L2.
// block = 256 threads = 8 waves; wave (wm = w>>1, wn = w&1) owns rows
// [mblock*64 + wm*16) x cols [wn*64, 4 x 16) -> 4 v8f accumulators.
// Split-fp32 GEMM: acc += Ahi*Bhi + Ahi*Blo + Alo*Bhi  (bf16 WMMA, f32 acc).
// Pipeline: reg-buffered staging + double-buffered LDS, 1 barrier per K-step:
//   load(ks+1)->regs | wmma(ks) from LDS[cur] | cvt+store(ks+1)->LDS[cur^1]
// Epilogue: out = relu(h_self + prelu(acc + bias)).
// ---------------------------------------------------------------------------
__global__ __launch_bounds__(256, 2)
void crf_iter(const float* __restrict__ h0, const float* __restrict__ h1,
              const float* __restrict__ h2,
              long bs0, long bs1, long bs2,          // batch strides (floats)
              float* __restrict__ outBase, long outBStride,
              const __bf16* __restrict__ aprep,
              const float* __restrict__ biasBase,
              const float* __restrict__ preluA)
{
    // branch constants
    const int Cs[3]      = {228, 111, 51};
    const int K1s[3]     = {111, 228, 228};
    const int K2s[3]     = {51, 51, 111};
    const int KstepsA[3] = {6, 9, 11};
    const int recOff[3]  = {0, 96, 168};
    const int biasOff[3] = {0, 256, 384};
    const int chOff[3]   = {0, 228, 339};

    int y = blockIdx.x;                  // 0..6 : (branch, mblock)
    int br, mblock;
    if (y < 4)       { br = 0; mblock = y; }
    else if (y < 6)  { br = 1; mblock = y - 4; }
    else             { br = 2; mblock = y - 6; }

    const float* srcA; const float* srcB; const float* srcSelf;
    long bsA, bsB, bsSelf;
    if (br == 0)      { srcA = h1; bsA = bs1; srcB = h2; bsB = bs2; srcSelf = h0; bsSelf = bs0; }
    else if (br == 1) { srcA = h0; bsA = bs0; srcB = h2; bsB = bs2; srcSelf = h1; bsSelf = bs1; }
    else              { srcA = h0; bsA = bs0; srcB = h1; bsB = bs1; srcSelf = h2; bsSelf = bs2; }

    const int C      = Cs[br];
    const int K1     = K1s[br];
    const int K2     = K2s[br];
    const int Ksteps = KstepsA[br];

    const int b     = blockIdx.z;
    const int nbase = blockIdx.y * 128;

    const int tid  = threadIdx.x;
    const int lane = tid & 31;
    const int wave = tid >> 5;
    const int wm   = wave >> 1;   // 0..3  (16-row subtile)
    const int wn   = wave & 1;    // 0..1  (64-col half)
    const int half = lane >> 4;

    // wave-uniform high bit of tid -> K-quad index is scalar (SGPR) so the
    // source-selection branches in the staging loads are s_cbranch, not exec-mask.
    const int tidHigh = __builtin_amdgcn_readfirstlane(tid >> 7);
    const int n       = tid & 127;   // pixel within tile for staging

    // Double-buffered LDS: activation tile transposed to [N=128][K=32] bf16,
    // row stride 40 halfs (80 B): packed b64 quad stores are <=2-way, b128
    // fragment loads hit the inherent 2-pass minimum and stay 16B-aligned.
    __shared__ __bf16 ldsHi[2][128 * 40];
    __shared__ __bf16 ldsLo[2][128 * 40];

    v8f acc[4] = {};

    const __bf16* arecBase =
        aprep + (size_t)(recOff[br] + (mblock * 4 + wm) * Ksteps) * 1024;

    const float* srcAb = srcA + (size_t)b * bsA + nbase;
    const float* srcBb = srcB + (size_t)b * bsB + nbase;

    // staging registers: 4 K-quads x 4 rows
    float stg[16];

    // --- loadStage: 16 coalesced b32 loads for K-step ks (scalar row select) ---
    auto loadStage = [&](int ks) {
#pragma unroll
        for (int it = 0; it < 4; ++it) {
            const int kq = it * 2 + tidHigh;   // scalar 0..7
            const int kl = kq * 4;             // local K base 0..28
#pragma unroll
            for (int j = 0; j < 4; ++j) {
                int kg = ks * 32 + kl + j;     // scalar row index
                float v = 0.0f;
                if (kg < K1)            v = srcAb[(size_t)kg * HW + n];
                else if (kg < K1 + K2)  v = srcBb[(size_t)(kg - K1) * HW + n];
                stg[it * 4 + j] = v;
            }
        }
    };

    // --- storeStage: branch-free hi/lo split + packed b64 stores ---
    auto storeStage = [&](__bf16* dHi, __bf16* dLo) {
#pragma unroll
        for (int it = 0; it < 4; ++it) {
            const int kq = it * 2 + tidHigh;
            const int kl = kq * 4;
            union { __bf16 h[4]; unsigned int u[2]; } ph, pl;
#pragma unroll
            for (int j = 0; j < 4; ++j) {
                float v = stg[it * 4 + j];
                __bf16 hv = (__bf16)v;
                ph.h[j] = hv;
                pl.h[j] = (__bf16)(v - (float)hv);
            }
            const int idx = n * 40 + kl;       // halfs; byte addr 8-aligned
            *reinterpret_cast<uint2*>(&dHi[idx]) =
                *reinterpret_cast<const uint2*>(ph.u);
            *reinterpret_cast<uint2*>(&dLo[idx]) =
                *reinterpret_cast<const uint2*>(pl.u);
        }
    };

    // prologue: fill buffer 0
    loadStage(0);
    storeStage(ldsHi[0], ldsLo[0]);
    __syncthreads();

    int cur = 0;
    for (int ks = 0; ks < Ksteps; ++ks) {
        const bool more = (ks + 1 < Ksteps);
        // 1) issue next K-step's global loads early: latency hides under WMMAs
        if (more) loadStage(ks + 1);

        // 2) A fragments (hi, lo) straight from pre-swizzled global
        const __bf16* ap = arecBase + (size_t)ks * 1024 + (size_t)lane * 32;
        FragBF ahi, alo;
        ahi.h[0] = *(const v8bf*)(ap);
        ahi.h[1] = *(const v8bf*)(ap + 8);
        alo.h[0] = *(const v8bf*)(ap + 16);
        alo.h[1] = *(const v8bf*)(ap + 24);

        // 3) 12 WMMAs from current LDS buffer
        const __bf16* bHi = ldsHi[cur];
        const __bf16* bLo = ldsLo[cur];
        const int nrow = wn * 64 + (lane & 15);
#pragma unroll
        for (int t = 0; t < 4; ++t) {
            const __bf16* bph = &bHi[(nrow + t * 16) * 40 + half * 8];
            const __bf16* bpl = &bLo[(nrow + t * 16) * 40 + half * 8];
            FragBF bhi, blo;
            bhi.h[0] = *(const v8bf*)(bph);
            bhi.h[1] = *(const v8bf*)(bph + 16);
            blo.h[0] = *(const v8bf*)(bpl);
            blo.h[1] = *(const v8bf*)(bpl + 16);

            acc[t] = __builtin_amdgcn_wmma_f32_16x16x32_bf16(
                false, ahi.v, false, bhi.v, (short)0, acc[t], false, false);
            acc[t] = __builtin_amdgcn_wmma_f32_16x16x32_bf16(
                false, ahi.v, false, blo.v, (short)0, acc[t], false, false);
            acc[t] = __builtin_amdgcn_wmma_f32_16x16x32_bf16(
                false, alo.v, false, bhi.v, (short)0, acc[t], false, false);
        }

        // 4) convert + store next tile into the other buffer, single barrier
        if (more) storeStage(ldsHi[cur ^ 1], ldsLo[cur ^ 1]);
        __syncthreads();
        cur ^= 1;
    }

    // --- epilogue: bias + prelu + skip-add + relu, concat-layout store ---
    const float  a_p   = preluA[0];
    const float* selfP = srcSelf + (size_t)b * bsSelf;
    float*       outP  = outBase + (size_t)b * outBStride + (size_t)chOff[br] * HW;
    const float* bb    = biasBase + biasOff[br];
    const int mrowBase = mblock * 64 + wm * 16 + half * 8;

#pragma unroll
    for (int t = 0; t < 4; ++t) {
        int col = nbase + wn * 64 + t * 16 + (lane & 15);
#pragma unroll
        for (int v = 0; v < 8; ++v) {
            int Mrow = mrowBase + v;
            if (Mrow < C) {
                float binary = acc[t][v] + bb[Mrow];
                float p  = binary >= 0.0f ? binary : a_p * binary;
                float hv = selfP[(size_t)Mrow * HW + col];
                float o  = hv + p;
                outP[(size_t)Mrow * HW + col] = o > 0.0f ? o : 0.0f;
            }
        }
    }
}

// ---------------------------------------------------------------------------
extern "C" void kernel_launch(void* const* d_in, const int* in_sizes, int n_in,
                              void* d_out, int out_size, void* d_ws, size_t ws_size,
                              hipStream_t stream) {
    const float* x0    = (const float*)d_in[0];
    const float* x1    = (const float*)d_in[1];
    const float* x2    = (const float*)d_in[2];
    const float* w_0_1 = (const float*)d_in[3];
    const float* b_0_1 = (const float*)d_in[4];
    const float* w_0_2 = (const float*)d_in[5];
    const float* b_0_2 = (const float*)d_in[6];
    const float* w_1_0 = (const float*)d_in[7];
    const float* b_1_0 = (const float*)d_in[8];
    const float* w_1_2 = (const float*)d_in[9];
    const float* b_1_2 = (const float*)d_in[10];
    const float* w_2_0 = (const float*)d_in[11];
    const float* b_2_0 = (const float*)d_in[12];
    const float* w_2_1 = (const float*)d_in[13];
    const float* b_2_1 = (const float*)d_in[14];
    const float* prelu = (const float*)d_in[15];

    // workspace layout
    float* hmid = (float*)d_ws;                                   // [2][390][HW] fp32
    size_t hmidBytes  = (size_t)2 * NCH_TOT * HW * sizeof(float); // 204,472,320 B
    __bf16* aprep = (__bf16*)((char*)d_ws + hmidBytes);           // 212 recs * 2048 B
    size_t aprepBytes = (size_t)212 * 2048;
    float* bias = (float*)((char*)d_ws + hmidBytes + aprepBytes); // 448 floats

    // --- prologue: weight swizzle (hi/lo bf16) + bias sums ---
    prep_weights<<<dim3(96), dim3(32), 0, stream>>>(w_1_0, w_2_0, 228, 111, 51, 6,
                                                    aprep + 0);
    prep_weights<<<dim3(72), dim3(32), 0, stream>>>(w_0_1, w_2_1, 111, 228, 51, 9,
                                                    aprep + (size_t)96 * 1024);
    prep_weights<<<dim3(44), dim3(32), 0, stream>>>(w_0_2, w_1_2, 51, 228, 111, 11,
                                                    aprep + (size_t)168 * 1024);
    prep_bias<<<dim3(1), dim3(256), 0, stream>>>(b_1_0, b_2_0, 228, bias + 0);
    prep_bias<<<dim3(1), dim3(128), 0, stream>>>(b_0_1, b_2_1, 111, bias + 256);
    prep_bias<<<dim3(1), dim3(64),  0, stream>>>(b_0_2, b_1_2,  51, bias + 384);

    dim3 grid(7, 512, 2);
    dim3 block(256);

    // iteration 1: sources = inputs, dest = hmid (concat layout)
    crf_iter<<<grid, block, 0, stream>>>(
        x0, x1, x2,
        (long)228 * HW, (long)111 * HW, (long)51 * HW,
        hmid, (long)NCH_TOT * HW,
        aprep, bias, prelu);

    // iteration 2: sources = hmid slices, dest = d_out (concat layout)
    crf_iter<<<grid, block, 0, stream>>>(
        hmid, hmid + (size_t)228 * HW, hmid + (size_t)339 * HW,
        (long)NCH_TOT * HW, (long)NCH_TOT * HW, (long)NCH_TOT * HW,
        (float*)d_out, (long)NCH_TOT * HW,
        aprep, bias, prelu);
}
